// SelfAttention_14044543057959
// MI455X (gfx1250) — matrix operand order repaired and checked by reading, not compiled, not run on hardware
//
#include <hip/hip_runtime.h>

// ---- problem constants (reference: B=8, N=2048, D=512, fp32 in/out) ----
#define BATCH 8
#define SEQ   2048
#define DIM   512
#define BR    64          // query rows per workgroup
#define BC    64          // keys per streamed tile
#define NT    (SEQ / BC)  // 32 K/V tiles
#define QSTR  520         // bf16-elem row stride for Qs / Ks (64-bank padding)
#define VSTR  72          // bf16-elem row stride for Vt / Ps
#define SSTR  68          // f32-elem row stride for Ss

typedef unsigned short ushort_t;
typedef ushort_t v8u  __attribute__((ext_vector_type(8)));
typedef ushort_t v16u __attribute__((ext_vector_type(16)));
typedef __bf16   v16bf __attribute__((ext_vector_type(16)));
typedef float    v8f  __attribute__((ext_vector_type(8)));
typedef unsigned int u32x4 __attribute__((ext_vector_type(4)));
typedef int i32x4 __attribute__((ext_vector_type(4)));
typedef int i32x8 __attribute__((ext_vector_type(8)));

// round-to-nearest-even f32 -> bf16
__device__ __forceinline__ ushort_t f2bf(float f) {
  unsigned u = __float_as_uint(f);
  u += 0x7FFFu + ((u >> 16) & 1u);
  return (ushort_t)(u >> 16);
}

// A-fragment (16x32 bf16): lane row fixed, K chunks {laneHi*8..+7, 16+laneHi*8..+7}
__device__ __forceinline__ v16u ldA(const ushort_t* rowptr, int laneHi) {
  const ushort_t* p = rowptr + laneHi * 8;
  v8u lo = *(const v8u*)p;
  v8u hi = *(const v8u*)(p + 16);
  return __builtin_shufflevector(lo, hi, 0,1,2,3,4,5,6,7,8,9,10,11,12,13,14,15);
}

// B-fragment (32x16 bf16): lane column fixed, K chunks {laneHi*8..+7, 16+laneHi*8..+7}
__device__ __forceinline__ v16u ldB(const ushort_t* rowptr, int laneHi) {
  const ushort_t* p = rowptr + laneHi * 8;
  v8u lo = *(const v8u*)p;
  v8u hi = *(const v8u*)(p + 16);
  return __builtin_shufflevector(lo, hi, 0,1,2,3,4,5,6,7,8,9,10,11,12,13,14,15);
}

__device__ __forceinline__ v8f bfwmma(v16u a, v16u b, v8f c) {
  return __builtin_amdgcn_wmma_f32_16x16x32_bf16(
      false, __builtin_bit_cast(v16bf, a),
      false, __builtin_bit_cast(v16bf, b),
      (short)0, c, false, false);
}

// ---- TDM: 2-D bf16 tile load, Global -> LDS, with LDS row padding ----
// D# packing per CDNA5 ISA §8.3/8.4. Dynamic LDS base offset is 0 (no static
// __shared__ in this kernel), so lds_addr = compile-time byte offsets below.
// pad_interval: DWORDs between pads = 2^(v+1); pad_amount: DWORDs added = v+1.
__device__ __forceinline__ void tdm_load_2d(unsigned lds_off, const ushort_t* gptr,
                                            unsigned tile_d0, unsigned tile_d1,
                                            unsigned stride0,
                                            unsigned pad_interval, unsigned pad_amount) {
  unsigned long long ga = (unsigned long long)(uintptr_t)gptr;
  u32x4 g0;
  g0.x = 1u;                                    // count=1, user descriptor
  g0.y = lds_off;                               // lds_addr (bytes)
  g0.z = (unsigned)(ga & 0xFFFFFFFFu);          // global_addr[31:0]
  g0.w = (unsigned)(ga >> 32) | (2u << 30);     // global_addr[56:32] | type=2
  i32x8 g1;
  g1[0] = (int)((1u << 16)                      // data_size = 2 bytes
              | (1u << 20)                      // pad_enable
              | (pad_interval << 22)
              | (pad_amount << 25));
  g1[1] = (int)((tile_d0 & 0xFFFFu) << 16);     // tensor_dim0[15:0] (== tile)
  g1[2] = (int)((tile_d0 >> 16) | ((tile_d1 & 0xFFFFu) << 16)); // d0 hi | tensor_dim1 lo
  g1[3] = (int)((tile_d1 >> 16) | (tile_d0 << 16));             // d1 hi | tile_dim0
  g1[4] = (int)tile_d1;                         // tile_dim1 | tile_dim2(=0)<<16
  g1[5] = (int)stride0;                         // tensor_dim0_stride[31:0]
  g1[6] = 0;                                    // stride0 hi | stride1 lo
  g1[7] = 0;
  i32x4 z4 = {0, 0, 0, 0};
#if defined(__clang_major__) && __clang_major__ >= 23
  i32x8 z8 = {0, 0, 0, 0, 0, 0, 0, 0};
  __builtin_amdgcn_tensor_load_to_lds(g0, g1, z4, z4, z8, 0);
#else
  __builtin_amdgcn_tensor_load_to_lds(g0, g1, z4, z4, 0);
#endif
}

// ---- dynamic LDS layout (bytes) : ~294 KB <= 320 KB/WGP ----
#define OFF_QS   0
#define OFF_KS0  (OFF_QS  + BR * QSTR * 2)
#define OFF_KS1  (OFF_KS0 + BC * QSTR * 2)
#define OFF_VT   (OFF_KS1 + BC * QSTR * 2)
#define OFF_SS   (OFF_VT  + DIM * VSTR * 2)
#define OFF_PS   (OFF_SS  + BR * SSTR * 4)
#define OFF_ST   (OFF_PS  + BR * VSTR * 2)
#define SMEM_BYTES (OFF_ST + 3 * BR * 4)

// ===================== pre-pass: f32 -> bf16 (+ transpose) =====================
__global__ __launch_bounds__(256)
void convert_bf16_kernel(const float* __restrict__ x,
                         ushort_t* __restrict__ xbf,    // [B][SEQ][DIM]
                         ushort_t* __restrict__ xbft) { // [B][DIM][SEQ]
  __shared__ float tile[64][65];
  const int bb = blockIdx.z;
  const int n0 = blockIdx.x * 64;   // SEQ tile
  const int d0 = blockIdx.y * 64;   // DIM tile
  const int tid = threadIdx.x;
  const float* xb = x + (size_t)bb * SEQ * DIM;

  #pragma unroll
  for (int it = 0; it < 4; ++it) {
    int idx = (tid + it * 256) * 4;
    int r = idx >> 6, c = idx & 63;
    float4 v = *(const float4*)(xb + (size_t)(n0 + r) * DIM + d0 + c);
    tile[r][c] = v.x; tile[r][c+1] = v.y; tile[r][c+2] = v.z; tile[r][c+3] = v.w;
  }
  __syncthreads();

  ushort_t* orow = xbf + ((size_t)bb * SEQ + n0) * DIM + d0;
  #pragma unroll
  for (int it = 0; it < 4; ++it) {
    int idx = (tid + it * 256) * 4;
    int r = idx >> 6, c = idx & 63;
    unsigned p01 = (unsigned)f2bf(tile[r][c])   | ((unsigned)f2bf(tile[r][c+1]) << 16);
    unsigned p23 = (unsigned)f2bf(tile[r][c+2]) | ((unsigned)f2bf(tile[r][c+3]) << 16);
    *(unsigned*)&orow[(size_t)r * DIM + c]     = p01;
    *(unsigned*)&orow[(size_t)r * DIM + c + 2] = p23;
  }
  ushort_t* ocol = xbft + ((size_t)bb * DIM + d0) * SEQ + n0;
  #pragma unroll
  for (int it = 0; it < 4; ++it) {
    int idx = (tid + it * 256) * 4;
    int r = idx >> 6, c = idx & 63;   // r: dim, c: seq
    unsigned p01 = (unsigned)f2bf(tile[c][r])   | ((unsigned)f2bf(tile[c+1][r]) << 16);
    unsigned p23 = (unsigned)f2bf(tile[c+2][r]) | ((unsigned)f2bf(tile[c+3][r]) << 16);
    *(unsigned*)&ocol[(size_t)r * SEQ + c]     = p01;
    *(unsigned*)&ocol[(size_t)r * SEQ + c + 2] = p23;
  }
}

// ===================== flash attention, bf16 WMMA + TDM =====================
__global__ __launch_bounds__(256, 1)
void flash_attn_bf16_kernel(const ushort_t* __restrict__ xbf,
                            const ushort_t* __restrict__ xbft,
                            float* __restrict__ out) {
  extern __shared__ char smem_raw[];
  ushort_t* Qs   = (ushort_t*)(smem_raw + OFF_QS);   // [BR][QSTR]
  ushort_t* Ks0  = (ushort_t*)(smem_raw + OFF_KS0);  // [BC][QSTR] x2 (double buffer)
  ushort_t* Vt   = (ushort_t*)(smem_raw + OFF_VT);   // [DIM][VSTR] (V transposed)
  float*    Ss   = (float*)   (smem_raw + OFF_SS);   // [BR][SSTR]
  ushort_t* Ps   = (ushort_t*)(smem_raw + OFF_PS);   // [BR][VSTR]
  float*    mrow = (float*)   (smem_raw + OFF_ST);
  float*    lrow = mrow + BR;
  float*    rsc  = lrow + BR;

  const int tid    = threadIdx.x;
  const int wave   = tid >> 5;
  const int lane   = tid & 31;
  const int laneHi = lane >> 4;
  const int lane16 = lane & 15;
  const int bidx   = blockIdx.y;
  const int r0     = blockIdx.x * BR;
  const ushort_t* xb  = xbf  + (size_t)bidx * SEQ * DIM;   // rows:  [SEQ][DIM]
  const ushort_t* xbt = xbft + (size_t)bidx * DIM * SEQ;   // cols:  [DIM][SEQ]

  if (tid < BR) { mrow[tid] = -__builtin_inff(); lrow[tid] = 0.f; }

  // prologue: TDM Q tile + K0 + V0 (pad makes LDS strides 520 / 72)
  if (wave == 0) {
    tdm_load_2d(OFF_QS,  xb + (size_t)r0 * DIM, DIM, BR, DIM, 7u, 3u); // 256DW rows, +4DW pad
    tdm_load_2d(OFF_KS0, xb,                    DIM, BC, DIM, 7u, 3u);
    tdm_load_2d(OFF_VT,  xbt,                   BC, DIM, SEQ, 4u, 3u); // 32DW rows, +4DW pad
    __builtin_amdgcn_s_wait_tensorcnt(1);  // Q + K0 landed; V0 still in flight
  }
  __syncthreads();

  const int rg = wave >> 1;   // O row group (16 rows)
  const int ch = wave & 1;    // O column half (256 cols)
  v8f o[16];
  v8f vzero = {};
  #pragma unroll
  for (int t = 0; t < 16; ++t) o[t] = vzero;

  for (int kt = 0; kt < NT; ++kt) {
    const int buf = kt & 1;
    const ushort_t* Ks = Ks0 + (size_t)buf * (BC * QSTR);  // no LDS-pointer array

    // ---- S = Q * K^T : 16 tiles, 2 per wave, 16 K-steps ----
    #pragma unroll
    for (int tt = 0; tt < 2; ++tt) {
      int t  = wave * 2 + tt;
      int mi = t >> 2, ni = t & 3;
      v8f acc = vzero;
      const ushort_t* Ab = Qs + (mi * 16 + lane16) * QSTR;
      const ushort_t* Bb = Ks + (ni * 16 + lane16) * QSTR; // B[k][n] = K[n][k]
      #pragma unroll
      for (int kc = 0; kc < DIM / 32; ++kc) {
        acc = bfwmma(ldA(Ab + kc * 32, laneHi), ldB(Bb + kc * 32, laneHi), acc);
      }
      #pragma unroll
      for (int r = 0; r < 8; ++r)
        Ss[(mi * 16 + laneHi * 8 + r) * SSTR + ni * 16 + lane16] = acc[r];
    }
    __syncthreads();                                   // (A) Ss ready, Ks[!buf] free

    // prefetch next K tile while softmax + PV run
    if (wave == 0 && kt + 1 < NT)
      tdm_load_2d(buf ? OFF_KS0 : OFF_KS1, xb + (size_t)(kt + 1) * BC * DIM,
                  DIM, BC, DIM, 7u, 3u);

    // ---- online softmax: 4 threads per row, shfl_xor butterflies ----
    {
      const int row = tid >> 2;
      const int q   = tid & 3;
      const float* sr = Ss + row * SSTR + q * 16;
      float mt = -__builtin_inff();
      #pragma unroll
      for (int c = 0; c < 16; ++c) mt = fmaxf(mt, sr[c]);
      mt = fmaxf(mt, __shfl_xor(mt, 1));
      mt = fmaxf(mt, __shfl_xor(mt, 2));
      float mold = mrow[row];
      float mnew = fmaxf(mold, mt);
      float s = 0.f;
      ushort_t* pr = Ps + row * VSTR + q * 16;
      #pragma unroll
      for (int c = 0; c < 16; c += 2) {
        float e0 = __expf(sr[c]     - mnew);
        float e1 = __expf(sr[c + 1] - mnew);
        s += e0 + e1;
        *(unsigned*)&pr[c] = (unsigned)f2bf(e0) | ((unsigned)f2bf(e1) << 16);
      }
      s += __shfl_xor(s, 1);
      s += __shfl_xor(s, 2);
      if (q == 0) {
        float scale = __expf(mold - mnew);
        lrow[row] = lrow[row] * scale + s;
        mrow[row] = mnew;
        rsc[row]  = scale;
      }
    }
    if (wave == 0) {                 // V(kt) landed (K(kt+1) may remain in flight)
      if (kt + 1 < NT) __builtin_amdgcn_s_wait_tensorcnt(1);
      else             __builtin_amdgcn_s_wait_tensorcnt(0);
    }
    __syncthreads();                                   // (B) Ps + Vt valid

    // ---- O = O*scale + P @ V ----
    float scv[8];
    #pragma unroll
    for (int r = 0; r < 8; ++r) scv[r] = rsc[rg * 16 + laneHi * 8 + r];
    #pragma unroll
    for (int t = 0; t < 16; ++t)
      #pragma unroll
      for (int r = 0; r < 8; ++r) o[t][r] *= scv[r];

    const ushort_t* Pb = Ps + (rg * 16 + lane16) * VSTR;
    v16u a0 = ldA(Pb,      laneHi);   // keys 0..31
    v16u a1 = ldA(Pb + 32, laneHi);   // keys 32..63
    #pragma unroll
    for (int t = 0; t < 16; ++t) {
      const ushort_t* Vb = Vt + (ch * 256 + t * 16 + lane16) * VSTR;
      o[t] = bfwmma(a0, ldB(Vb,      laneHi), o[t]);
      o[t] = bfwmma(a1, ldB(Vb + 32, laneHi), o[t]);
    }
    __syncthreads();                                   // (C) PV done, Vt free

    if (wave == 0 && kt + 1 < NT) {
      tdm_load_2d(OFF_VT, xbt + (size_t)(kt + 1) * BC, BC, DIM, SEQ, 4u, 3u);
      __builtin_amdgcn_s_wait_tensorcnt(1);            // K(kt+1) landed (in-order)
    }
    __syncthreads();                                   // (D) release S-phase kt+1
  }

  // ---- finalize: O /= l, store f32 ----
  float inv[8];
  #pragma unroll
  for (int r = 0; r < 8; ++r) inv[r] = 1.f / lrow[rg * 16 + laneHi * 8 + r];
  #pragma unroll
  for (int t = 0; t < 16; ++t) {
    int col = ch * 256 + t * 16 + lane16;
    #pragma unroll
    for (int r = 0; r < 8; ++r) {
      int row = r0 + rg * 16 + laneHi * 8 + r;
      out[((size_t)bidx * SEQ + row) * DIM + col] = o[t][r] * inv[r];
    }
  }
}

extern "C" void kernel_launch(void* const* d_in, const int* in_sizes, int n_in,
                              void* d_out, int out_size, void* d_ws, size_t ws_size,
                              hipStream_t stream) {
  const float* x = (const float*)d_in[0];
  float* out = (float*)d_out;
  (void)in_sizes; (void)n_in; (void)out_size; (void)ws_size;

  // workspace: bf16 copy (16 MB) + bf16 transposed copy (16 MB)
  ushort_t* xbf  = (ushort_t*)d_ws;
  ushort_t* xbft = xbf + (size_t)BATCH * SEQ * DIM;

  convert_bf16_kernel<<<dim3(SEQ / 64, DIM / 64, BATCH), 256, 0, stream>>>(x, xbf, xbft);

  (void)hipFuncSetAttribute((const void*)flash_attn_bf16_kernel,
                            hipFuncAttributeMaxDynamicSharedMemorySize, (int)SMEM_BYTES);
  flash_attn_bf16_kernel<<<dim3(SEQ / BR, BATCH), 256, SMEM_BYTES, stream>>>(xbf, xbft, out);
}
